// MMA_25821343383834
// MI455X (gfx1250) — matrix operand order repaired and checked by reading, not compiled, not run on hardware
//
#include <hip/hip_runtime.h>

typedef __attribute__((ext_vector_type(16))) __bf16 v16bf;
typedef __attribute__((ext_vector_type(8)))  float  v8f;
typedef __attribute__((ext_vector_type(4)))  unsigned int u32x4;
typedef __attribute__((ext_vector_type(8)))  int    i32x8;
typedef __attribute__((ext_vector_type(4)))  int    i32x4;
typedef __attribute__((ext_vector_type(4)))  float  f32x4;

#define BATCH 16
#define CCH   256
#define HWN   9216      // 96*96
#define NGRP  32

// padded LDS row stride for the TDM-staged 256x256 bf16 matrix:
// 512B rows + 16B hardware pad = 528B = 264 ushorts
#define MBROW 264
#define MBLDS (256 * MBROW)          // ushort count = 67584 (135168 B)
#define AWAVE (16 * 264 + 16 * 16)   // per-wave attn scratch (ushorts)

union Frag { u32x4 q[2]; v16bf v; };

__device__ __forceinline__ unsigned short f2bf(float f) {
    unsigned u = __builtin_bit_cast(unsigned, f);
    u += 0x7FFFu + ((u >> 16) & 1u);
    return (unsigned short)(u >> 16);
}

__device__ __forceinline__ v8f wmma_bf16(const Frag& a, const Frag& b, v8f c) {
    return __builtin_amdgcn_wmma_f32_16x16x32_bf16(false, a.v, false, b.v,
                                                   (short)0, c, false, false);
}

// ---------------------------------------------------------------------------
// TDM: stage a 256x256 bf16 row-major matrix (512B rows) into LDS at ldsoff,
// with 16B padding appended per 512B row (row stride 528B in LDS).
// D# per CDNA5 ISA 8.3/8.4: data_size=8B, tensor/tile 64x256 units,
// dim0_stride=64, pad_interval=128 dwords (code 6), pad_amount=4 dwords
// (code 3), 2-D tensor (groups 2/3 zero).  6-arg builtin form (clang-23).
// ---------------------------------------------------------------------------
__device__ __forceinline__ void tdm_load_mat(const unsigned short* gsrc,
                                             unsigned ldsoff) {
    unsigned long long ga = (unsigned long long)(size_t)gsrc;
    u32x4 g0;
    g0.x = 1u;                                             // count=1 (valid)
    g0.y = ldsoff;                                         // lds_addr (bytes)
    g0.z = (unsigned)(ga & 0xFFFFFFFFu);                   // global_addr lo
    g0.w = (unsigned)((ga >> 32) & 0x01FFFFFFu) | (2u << 30); // addr hi | type=2
    i32x8 g1;
    g1[0] = (3 << 16) | (1 << 20) | (6 << 22) | (3 << 25); // 8B elems, pad en,
                                                           // every 128 dw, +4 dw
    g1[1] = (64 << 16);            // tensor_dim0[15:0]=64 (row = 64 x 8B)
    g1[2] = (256 << 16);           // tensor_dim0 hi=0 | tensor_dim1 lo=256
    g1[3] = (64 << 16);            // tensor_dim1 hi=0 | tile_dim0=64
    g1[4] = 256;                   // tile_dim1=256 | tile_dim2=0
    g1[5] = 64;                    // tensor_dim0_stride lo = 64 units
    g1[6] = 0;                     // stride0 hi | stride1 lo
    g1[7] = 0;                     // stride1 hi
    i32x4 z4 = {0, 0, 0, 0};
    i32x8 z8 = {0, 0, 0, 0, 0, 0, 0, 0};
    __builtin_amdgcn_tensor_load_to_lds(g0, g1, z4, z4, z8, 0);
}

// ---------------------------------------------------------------------------
// Kernel P: convert weights to bf16 (+ mb transposed copy)
// ---------------------------------------------------------------------------
__global__ void prep_weights(const float* __restrict__ phi_w,
                             const float* __restrict__ mb,
                             const float* __restrict__ wz_w,
                             unsigned short* __restrict__ pw_bf,
                             unsigned short* __restrict__ mb_bf,
                             unsigned short* __restrict__ mbT_bf,
                             unsigned short* __restrict__ wz_bf) {
    int i = blockIdx.x * 256 + threadIdx.x;        // 0..65535
    float m = mb[i];
    pw_bf[i] = f2bf(phi_w[i]);
    wz_bf[i] = f2bf(wz_w[i]);
    mb_bf[i] = f2bf(m);
    int c = i >> 8, k = i & 255;
    mbT_bf[k * 256 + c] = f2bf(m);
}

// ---------------------------------------------------------------------------
// Kernel S: GroupNorm stats: one block per (b, group); 8 ch x 9216 px
// ---------------------------------------------------------------------------
__global__ void gn_stats(const float* __restrict__ x, float* __restrict__ stats) {
    int bg = blockIdx.x;                 // 0..511
    int b = bg >> 5, g = bg & 31;
    const float* xp = x + ((size_t)(b * CCH + g * 8)) * HWN;
    float s = 0.f, s2 = 0.f;
    for (int cc = 0; cc < 8; ++cc) {
        const float* row = xp + (size_t)cc * HWN;
        for (int n = threadIdx.x * 4; n < HWN; n += 1024) {
            f32x4 v = *(const f32x4*)(row + n);
            s  += v.x + v.y + v.z + v.w;
            s2 += v.x * v.x + v.y * v.y + v.z * v.z + v.w * v.w;
        }
    }
    __shared__ float rs[256], rq[256];
    int t = threadIdx.x;
    rs[t] = s; rq[t] = s2;
    __syncthreads();
    for (int off = 128; off > 0; off >>= 1) {
        if (t < off) { rs[t] += rs[t + off]; rq[t] += rq[t + off]; }
        __syncthreads();
    }
    if (t == 0) {
        float inv = 1.0f / (8.0f * HWN);
        float mean = rs[0] * inv;
        float var  = rq[0] * inv - mean * mean;
        stats[bg * 2]     = mean;
        stats[bg * 2 + 1] = rsqrtf(var + 1e-6f);
    }
}

// ---------------------------------------------------------------------------
// Kernel N: apply GroupNorm affine, convert to bf16, transpose to token-major
// qn[b][n][c].  64c x 64n tile through LDS.
// ---------------------------------------------------------------------------
__global__ void __launch_bounds__(256)
gn_apply_transpose(const float* __restrict__ x, const float* __restrict__ stats,
                   const float* __restrict__ gamma, const float* __restrict__ beta,
                   unsigned short* __restrict__ qn) {
    __shared__ unsigned short T[64][72];   // [n][c], padded
    int n0 = blockIdx.x * 64, c0 = blockIdx.y * 64, b = blockIdx.z;
    int t = threadIdx.x;
    int cc = t >> 2, seg = (t & 3) * 16;
    int c = c0 + cc, g = c >> 3;
    float mean = stats[(b * 32 + g) * 2];
    float rstd = stats[(b * 32 + g) * 2 + 1];
    float sc = gamma[c] * rstd;
    float sh = beta[c] - mean * sc;
    const float* row = x + ((size_t)(b * CCH + c)) * HWN + n0 + seg;
#pragma unroll
    for (int i = 0; i < 16; i += 4) {
        f32x4 v = *(const f32x4*)(row + i);
        T[seg + i + 0][cc] = f2bf(v.x * sc + sh);
        T[seg + i + 1][cc] = f2bf(v.y * sc + sh);
        T[seg + i + 2][cc] = f2bf(v.z * sc + sh);
        T[seg + i + 3][cc] = f2bf(v.w * sc + sh);
    }
    __syncthreads();
    int nn = t >> 2, cs = (t & 3) * 16;
    unsigned short* dst = qn + ((size_t)b * HWN + n0 + nn) * CCH + c0 + cs;
    const u32x4* src = (const u32x4*)&T[nn][cs];
    ((u32x4*)dst)[0] = src[0];
    ((u32x4*)dst)[1] = src[1];
}

// ---------------------------------------------------------------------------
// Kernel G1: phi = phi_w x qn (+phi_b).  Output token-major bf16 phi[b][n][c].
// WG = 128(o) x 128(n), 8 waves, direct global operand loads (weights L2-hot).
// ---------------------------------------------------------------------------
__global__ void __launch_bounds__(256)
gemm_phi(const unsigned short* __restrict__ W,    // [o][c] bf16
         const unsigned short* __restrict__ Xq,   // qn token-major
         const float* __restrict__ bias,          // phi_b
         unsigned short* __restrict__ Out) {      // phi token-major
    __shared__ unsigned short sOut[128][136];
    int b = blockIdx.z;
    int oblk = blockIdx.y * 128, nblk = blockIdx.x * 128;
    int tid = threadIdx.x, lane = tid & 31, wave = tid >> 5;
    int wo = wave >> 2, wn = wave & 3;
    int lrow = lane & 15, lhi = lane >> 4;
    const unsigned short* Xb = Xq + (size_t)b * HWN * CCH;

    v8f acc[4][2];
#pragma unroll
    for (int i = 0; i < 4; ++i)
#pragma unroll
        for (int j = 0; j < 2; ++j) acc[i][j] = (v8f)(0.0f);

    for (int k0 = 0; k0 < 256; k0 += 32) {
        Frag a[4], bf[2];
#pragma unroll
        for (int ti = 0; ti < 4; ++ti) {
            int r = oblk + wo * 64 + ti * 16 + lrow;
            const u32x4* p = (const u32x4*)(W + r * 256 + k0 + lhi * 16);
            a[ti].q[0] = p[0];
            a[ti].q[1] = p[1];          // +8 elements
        }
#pragma unroll
        for (int tj = 0; tj < 2; ++tj) {
            int tok = nblk + wn * 32 + tj * 16 + lrow;
            const u32x4* p = (const u32x4*)(Xb + (size_t)tok * 256 + k0 + lhi * 16);
            bf[tj].q[0] = p[0];
            bf[tj].q[1] = p[1];
            __builtin_prefetch(Xb + (size_t)tok * 256 + k0 + lhi * 16 + 64, 0, 1);
        }
#pragma unroll
        for (int ti = 0; ti < 4; ++ti)
#pragma unroll
            for (int tj = 0; tj < 2; ++tj)
                acc[ti][tj] = wmma_bf16(a[ti], bf[tj], acc[ti][tj]);
    }
#pragma unroll
    for (int ti = 0; ti < 4; ++ti)
#pragma unroll
        for (int tj = 0; tj < 2; ++tj)
#pragma unroll
            for (int r = 0; r < 8; ++r) {
                int o_l = wo * 64 + ti * 16 + r + 8 * lhi;
                int n_l = wn * 32 + tj * 16 + lrow;
                sOut[n_l][o_l] = f2bf(acc[ti][tj][r] + bias[oblk + o_l]);
            }
    __syncthreads();
    int n_l = tid >> 1, seg = (tid & 1) * 64;
    unsigned short* dst = Out + ((size_t)b * HWN + nblk + n_l) * CCH + oblk + seg;
    const u32x4* src = (const u32x4*)&sOut[n_l][seg];
#pragma unroll
    for (int q = 0; q < 8; ++q) ((u32x4*)dst)[q] = src[q];
}

// ---------------------------------------------------------------------------
// Kernel A: fused scores -> softmax -> y.  One wave = 16 tokens.
// mbT (phase 1) and mb (phase 2) are staged whole into LDS by the Tensor
// Data Mover with hardware row padding (528B stride -> conflict-free b128
// reads).  scores[16x256] kept in registers, softmax via shfl_xor within
// 16-lane halves, attn staged per-wave in LDS in A-operand layout.
// ---------------------------------------------------------------------------
__global__ void __launch_bounds__(256)
attn_fused(const unsigned short* __restrict__ Tok,   // phi token-major
           const unsigned short* __restrict__ MbT,   // [k][c]
           const unsigned short* __restrict__ Mb,    // [c][k]
           unsigned short* __restrict__ Y) {         // token-major
    extern __shared__ unsigned short dynsm[];
    unsigned short* sMat = dynsm;                    // TDM-staged matrix
    int b = blockIdx.y;
    int tid = threadIdx.x, lane = tid & 31, wave = tid >> 5;
    int lrow = lane & 15, lhi = lane >> 4;
    int n0 = (blockIdx.x * 8 + wave) * 16;
    unsigned short* sAttn = dynsm + MBLDS + wave * AWAVE;
    unsigned short* sMini = sAttn + 16 * 264;
    const unsigned short* tok = Tok + ((size_t)b * HWN + n0) * CCH;

    // kick off TDM load of mbT while all waves fetch their A fragments
    if (wave == 0) {
        tdm_load_mat(MbT, (unsigned)(size_t)sMat);
    }

    // preload A fragments (this wave's 16 token rows)
    Frag af[8];
#pragma unroll
    for (int kk = 0; kk < 8; ++kk) {
        const u32x4* p = (const u32x4*)(tok + (size_t)lrow * 256 + kk * 32 + lhi * 16);
        af[kk].q[0] = p[0];
        af[kk].q[1] = p[1];
    }

    v8f sacc[16];
#pragma unroll
    for (int kt = 0; kt < 16; ++kt) sacc[kt] = (v8f)(0.0f);

    if (wave == 0) __builtin_amdgcn_s_wait_tensorcnt(0);
    __syncthreads();

    for (int kt = 0; kt < 16; ++kt) {
#pragma unroll
        for (int kk = 0; kk < 8; ++kk) {
            Frag bb;
            const u32x4* p = (const u32x4*)(sMat + (kt * 16 + lrow) * MBROW + kk * 32 + lhi * 16);
            bb.q[0] = p[0];
            bb.q[1] = p[1];
            sacc[kt] = wmma_bf16(af[kk], bb, sacc[kt]);
        }
    }
    __syncthreads();                 // everyone done reading mbT
    if (wave == 0) {                 // refill the same buffer with mb
        tdm_load_mat(Mb, (unsigned)(size_t)sMat);
    }

    // softmax over 256 slots per row;  row M = r + 8*lhi lives in one 16-lane half
    const float scale = 0.0625f;     // 256^-0.5
    float mx[8], sm[8];
#pragma unroll
    for (int r = 0; r < 8; ++r) mx[r] = -3.4e38f;
#pragma unroll
    for (int kt = 0; kt < 16; ++kt)
#pragma unroll
        for (int r = 0; r < 8; ++r) mx[r] = fmaxf(mx[r], sacc[kt][r]);
#pragma unroll
    for (int r = 0; r < 8; ++r) {
        mx[r] = fmaxf(mx[r], __shfl_xor(mx[r], 1, 32));
        mx[r] = fmaxf(mx[r], __shfl_xor(mx[r], 2, 32));
        mx[r] = fmaxf(mx[r], __shfl_xor(mx[r], 4, 32));
        mx[r] = fmaxf(mx[r], __shfl_xor(mx[r], 8, 32));
    }
#pragma unroll
    for (int r = 0; r < 8; ++r) sm[r] = 0.f;
#pragma unroll
    for (int kt = 0; kt < 16; ++kt)
#pragma unroll
        for (int r = 0; r < 8; ++r) {
            float e = __expf((sacc[kt][r] - mx[r]) * scale);
            sacc[kt][r] = e;
            sm[r] += e;
        }
#pragma unroll
    for (int r = 0; r < 8; ++r) {
        sm[r] += __shfl_xor(sm[r], 1, 32);
        sm[r] += __shfl_xor(sm[r], 2, 32);
        sm[r] += __shfl_xor(sm[r], 4, 32);
        sm[r] += __shfl_xor(sm[r], 8, 32);
        sm[r] = 1.0f / sm[r];
    }
    // write attn (bf16) into per-wave LDS, row-major [16][264]
#pragma unroll
    for (int kt = 0; kt < 16; ++kt)
#pragma unroll
        for (int r = 0; r < 8; ++r)
            sAttn[(r + 8 * lhi) * 264 + kt * 16 + lrow] = f2bf(sacc[kt][r] * sm[r]);

    if (wave == 0) __builtin_amdgcn_s_wait_tensorcnt(0);
    __syncthreads();                 // mb staged + attn visible

    // y = attn x mb^T : A from per-wave LDS, B columns contiguous in staged mb
    for (int ct = 0; ct < 16; ++ct) {
        v8f acc2 = (v8f)(0.0f);
#pragma unroll
        for (int kk = 0; kk < 8; ++kk) {
            Frag a2, b2;
            const u32x4* pa = (const u32x4*)(sAttn + lrow * 264 + kk * 32 + lhi * 16);
            a2.q[0] = pa[0];
            a2.q[1] = pa[1];
            const u32x4* pb = (const u32x4*)(sMat + (ct * 16 + lrow) * MBROW + kk * 32 + lhi * 16);
            b2.q[0] = pb[0];
            b2.q[1] = pb[1];
            acc2 = wmma_bf16(a2, b2, acc2);
        }
        // stage 16x16 tile for coalesced token-major store
#pragma unroll
        for (int r = 0; r < 8; ++r)
            sMini[(r + 8 * lhi) * 16 + lrow] = f2bf(acc2[r]);
        int m = lane >> 1, h = lane & 1;
        u32x4 val = *(const u32x4*)(sMini + m * 16 + h * 8);
        *(u32x4*)(Y + ((size_t)b * HWN + n0 + m) * CCH + ct * 16 + h * 8) = val;
    }
}

// ---------------------------------------------------------------------------
// Kernel G2: out = wz_w x y + wz_b + x  (f32 output, residual fused)
// ---------------------------------------------------------------------------
__global__ void __launch_bounds__(256)
gemm_out(const unsigned short* __restrict__ W,    // wz bf16 [o][c]
         const unsigned short* __restrict__ Yt,   // y token-major
         const float* __restrict__ bias,          // wz_b
         const float* __restrict__ X,             // residual
         float* __restrict__ Out) {
    int b = blockIdx.z;
    int oblk = blockIdx.y * 128, nblk = blockIdx.x * 128;
    int tid = threadIdx.x, lane = tid & 31, wave = tid >> 5;
    int wo = wave >> 2, wn = wave & 3;
    int lrow = lane & 15, lhi = lane >> 4;
    const unsigned short* Yb = Yt + (size_t)b * HWN * CCH;

    v8f acc[4][2];
#pragma unroll
    for (int i = 0; i < 4; ++i)
#pragma unroll
        for (int j = 0; j < 2; ++j) acc[i][j] = (v8f)(0.0f);

    for (int k0 = 0; k0 < 256; k0 += 32) {
        Frag a[4], bf[2];
#pragma unroll
        for (int ti = 0; ti < 4; ++ti) {
            int r = oblk + wo * 64 + ti * 16 + lrow;
            const u32x4* p = (const u32x4*)(W + r * 256 + k0 + lhi * 16);
            a[ti].q[0] = p[0];
            a[ti].q[1] = p[1];
        }
#pragma unroll
        for (int tj = 0; tj < 2; ++tj) {
            int tok = nblk + wn * 32 + tj * 16 + lrow;
            const u32x4* p = (const u32x4*)(Yb + (size_t)tok * 256 + k0 + lhi * 16);
            bf[tj].q[0] = p[0];
            bf[tj].q[1] = p[1];
            __builtin_prefetch(Yb + (size_t)tok * 256 + k0 + lhi * 16 + 64, 0, 1);
        }
#pragma unroll
        for (int ti = 0; ti < 4; ++ti)
#pragma unroll
            for (int tj = 0; tj < 2; ++tj)
                acc[ti][tj] = wmma_bf16(a[ti], bf[tj], acc[ti][tj]);
    }
#pragma unroll
    for (int ti = 0; ti < 4; ++ti)
#pragma unroll
        for (int tj = 0; tj < 2; ++tj) {
            int n = nblk + wn * 32 + tj * 16 + lrow;
#pragma unroll
            for (int r = 0; r < 8; ++r) {
                int o = oblk + wo * 64 + ti * 16 + r + 8 * lhi;
                size_t idx = ((size_t)(b * CCH + o)) * HWN + n;
                Out[idx] = acc[ti][tj][r] + bias[o] + X[idx];
            }
        }
}

// ---------------------------------------------------------------------------
extern "C" void kernel_launch(void* const* d_in, const int* in_sizes, int n_in,
                              void* d_out, int out_size, void* d_ws, size_t ws_size,
                              hipStream_t stream) {
    const float* x     = (const float*)d_in[0];
    const float* gamma = (const float*)d_in[1];
    const float* beta  = (const float*)d_in[2];
    const float* phi_w = (const float*)d_in[3];
    const float* phi_b = (const float*)d_in[4];
    const float* mb    = (const float*)d_in[5];
    const float* wz_w  = (const float*)d_in[6];
    const float* wz_b  = (const float*)d_in[7];
    float* out = (float*)d_out;

    char* ws = (char*)d_ws;
    float*          stats  = (float*)(ws);                    // 4 KB
    unsigned short* pw_bf  = (unsigned short*)(ws + 4096);
    unsigned short* wz_bf  = pw_bf  + 65536;
    unsigned short* mb_bf  = wz_bf  + 65536;
    unsigned short* mbT_bf = mb_bf  + 65536;
    unsigned short* qn     = mbT_bf + 65536;                  // 75.5 MB
    unsigned short* phi    = qn  + (size_t)BATCH * HWN * CCH; // 75.5 MB
    unsigned short* yb     = phi + (size_t)BATCH * HWN * CCH; // 75.5 MB

    size_t dlds = (size_t)(MBLDS + 8 * AWAVE) * sizeof(unsigned short); // 206848 B
    (void)hipFuncSetAttribute((const void*)attn_fused,
                              hipFuncAttributeMaxDynamicSharedMemorySize, (int)dlds);

    prep_weights<<<256, 256, 0, stream>>>(phi_w, mb, wz_w, pw_bf, mb_bf, mbT_bf, wz_bf);
    gn_stats<<<512, 256, 0, stream>>>(x, stats);
    gn_apply_transpose<<<dim3(144, 4, 16), 256, 0, stream>>>(x, stats, gamma, beta, qn);
    gemm_phi<<<dim3(72, 2, 16), 256, 0, stream>>>(pw_bf, qn, phi_b, phi);
    attn_fused<<<dim3(72, 16), 256, dlds, stream>>>(phi, mbT_bf, mb_bf, yb);
    gemm_out<<<dim3(72, 2, 16), 256, 0, stream>>>(wz_bf, yb, wz_b, x, out);
}
